// GeneralAttention_24541443130171
// MI455X (gfx1250) — matrix operand order repaired and checked by reading, not compiled, not run on hardware
//
#include <hip/hip_runtime.h>
#include <hip/hip_bf16.h>

// ---------------------------------------------------------------------------
// GeneralAttention (B=8, TQ=TK=2048, DQ=DK=512), fp32 in / fp32 out.
//   qw = q @ W ; scores = qw @ k^T ; mask -> -inf ; P = softmax ; out = P @ v
// Outputs (concat): out (8*2048*512 f32), attention (8*2048*2048 f32).
//
// MI455X (gfx1250, wave32) strategy:
//  * All 3 GEMMs on v_wmma_f32_16x16x32_bf16 (fp32 WMMA only has K=4).
//  * One-time conversions: q,k -> bf16 ; v -> v^T bf16 ; W -> W^T bf16, so
//    every GEMM sees A row-major (K contiguous) and B^T row-major
//    (K contiguous per column) — the native WMMA fragment layouts; the
//    scores GEMM (qw x k^T) is NT and needs no transpose at all.
//  * attention is a required output -> it must be materialized; 2-pass
//    softmax over the stored scores adds no asymptotic traffic.
//  * GEMM epilogues are compile-time specialized (mask / f32 store / bf16
//    store) so they are branch-free and the mask loads can be clause'd.
//  * Workspace: qb 16.78M | kb 16.78M | vtb 16.78M | wtb 0.52M |
//               qwb 16.78M | pb 67.1M  => ~134.7 MB in d_ws.
// ---------------------------------------------------------------------------

typedef __attribute__((ext_vector_type(16))) __bf16 v16bf;
typedef __attribute__((ext_vector_type(8)))  __bf16 bf16x8;
typedef __attribute__((ext_vector_type(8)))  float  v8f;

#define ATT_B  8
#define ATT_TQ 2048
#define ATT_TK 2048
#define ATT_D  512

// ---------------- conversion kernels ----------------

__global__ void cvt_f32_to_bf16(const float* __restrict__ in,
                                __bf16* __restrict__ out, long n) {
  for (long i = (long)blockIdx.x * blockDim.x + threadIdx.x; i < n;
       i += (long)gridDim.x * blockDim.x)
    out[i] = (__bf16)in[i];
}

// out[b][c][r] = (bf16) in[b][r][c]
__global__ void transpose_f32_to_bf16(const float* __restrict__ in,
                                      __bf16* __restrict__ out,
                                      int rows, int cols) {
  long base = (long)blockIdx.z * rows * cols;
  long idx  = (long)blockIdx.x * blockDim.x + threadIdx.x;
  if (idx >= (long)rows * cols) return;
  int r = (int)(idx / cols);
  int c = (int)(idx % cols);
  out[base + (long)c * rows + r] = (__bf16)in[base + idx];
}

// ---------------- generic bf16 WMMA GEMM ----------------
// C[b] (MxN) = A[b] (MxK row-major) * BT[b]^T, BT row-major [n][k].
// 256 threads = 8 waves; block tile 128x128; wave tile 32x64
// (2x4 v_wmma_f32_16x16x32_bf16 per K-step of 32).
template <bool HAS_MASK, bool STORE_F32, bool STORE_BF16>
__global__ __launch_bounds__(256)
void wmma_gemm_bf16(const __bf16* __restrict__ A,
                    const __bf16* __restrict__ BT,
                    float* __restrict__ C,
                    __bf16* __restrict__ Cb,
                    const int* __restrict__ mask,
                    int K, int lda, int ldb, int ldc,
                    long long strideA, long long strideB,
                    long long strideC, long long strideM) {
  const int tid   = threadIdx.x;
  const int lane  = tid & 31;
  const int wave  = tid >> 5;
  const int waveM = wave >> 1;   // 0..3
  const int waveN = wave & 1;    // 0..1

  const int m0 = blockIdx.y * 128 + waveM * 32;  // wave M origin
  const int n0 = blockIdx.x * 128 + waveN * 64;  // wave N origin

  // Per-batch base pointers (hoisted 64-bit math; 32-bit offsets below).
  const __bf16* Ab = A  + (long)blockIdx.z * strideA;
  const __bf16* Bb = BT + (long)blockIdx.z * strideB;
  float*        Cf = STORE_F32  ? C  + (long)blockIdx.z * strideC : nullptr;
  __bf16*       Cg = STORE_BF16 ? Cb + (long)blockIdx.z * strideC : nullptr;
  const int*    Mb = HAS_MASK   ? mask + (long)blockIdx.z * strideM : nullptr;

  // A fragment (16x32 bf16), this code's contiguous k order (not the ISA table's; B matches):
  //   lane<16: M=lane,    K = 0..15
  //   lane>=16:M=lane-16, K = 16..31
  const int aM   = lane & 15;
  const int aKof = (lane >> 4) * 16;
  // B fragment (32x16 bf16): N = lane&15; lanes 0-15 K=0..15,
  // lanes 16-31 K=16..31 (contiguous per lane from BT row-major).
  const int bN   = lane & 15;
  const int bKof = (lane >> 4) * 16;

  v8f acc[2][4];
#pragma unroll
  for (int mi = 0; mi < 2; ++mi)
#pragma unroll
    for (int ni = 0; ni < 4; ++ni) acc[mi][ni] = (v8f){};

  for (int k0 = 0; k0 < K; k0 += 32) {
    v16bf afr[2];
#pragma unroll
    for (int mi = 0; mi < 2; ++mi) {
      const __bf16* p = Ab + (unsigned)((m0 + mi * 16 + aM) * lda + k0);
      bf16x8 lo = *(const bf16x8*)(p + aKof);
      bf16x8 hi = *(const bf16x8*)(p + 8 + aKof);
#pragma unroll
      for (int t = 0; t < 8; ++t) { afr[mi][t] = lo[t]; afr[mi][t + 8] = hi[t]; }
      __builtin_prefetch(p + 32, 0, 1);          // global_prefetch_b8
    }
    v16bf bfr[4];
#pragma unroll
    for (int ni = 0; ni < 4; ++ni) {
      const __bf16* p = Bb + (unsigned)((n0 + ni * 16 + bN) * ldb + k0 + bKof);
      bfr[ni] = *(const v16bf*)p;
      __builtin_prefetch(p + 32, 0, 1);
    }
#pragma unroll
    for (int mi = 0; mi < 2; ++mi)
#pragma unroll
      for (int ni = 0; ni < 4; ++ni)
        acc[mi][ni] = __builtin_amdgcn_wmma_f32_16x16x32_bf16(
            false, afr[mi], false, bfr[ni], (short)0, acc[mi][ni],
            false, false);
  }

  // C/D layout: VGPR r -> M = r + 8*(lane>=16), N = lane&15
  const int   cN = lane & 15;
  const int   cM = (lane >> 4) * 8;
  const float NEG_INF = -__builtin_inff();
#pragma unroll
  for (int mi = 0; mi < 2; ++mi)
#pragma unroll
    for (int ni = 0; ni < 4; ++ni) {
      const int row0 = m0 + mi * 16 + cM;
      const int col  = n0 + ni * 16 + cN;
      // Issue all mask loads for this tile first (branch-free, clause-able).
      int mv[8];
      if (HAS_MASK) {
#pragma unroll
        for (int r = 0; r < 8; ++r)
          mv[r] = Mb[(unsigned)((row0 + r) * ldc + col)];
      }
#pragma unroll
      for (int r = 0; r < 8; ++r) {
        const unsigned idx = (unsigned)((row0 + r) * ldc + col);
        float val = acc[mi][ni][r];
        if (HAS_MASK)   val = mv[r] ? NEG_INF : val;
        if (STORE_F32)  Cf[idx] = val;
        if (STORE_BF16) Cg[idx] = (__bf16)val;
      }
    }
}

// ---------------- row softmax (2048 cols, 256 threads/row) ----------------
__global__ __launch_bounds__(256)
void softmax_rows(float* __restrict__ S, __bf16* __restrict__ P) {
  const int row = blockIdx.x;
  float*  s = S + (long)row * ATT_TK;
  __bf16* p = P + (long)row * ATT_TK;
  __shared__ float red[256];
  const int tid = threadIdx.x;
  float vals[8];
  float m = -__builtin_inff();
#pragma unroll
  for (int i = 0; i < 8; ++i) {
    vals[i] = s[tid + i * 256];
    m = fmaxf(m, vals[i]);
  }
  red[tid] = m; __syncthreads();
  for (int off = 128; off > 0; off >>= 1) {
    if (tid < off) red[tid] = fmaxf(red[tid], red[tid + off]);
    __syncthreads();
  }
  m = red[0]; __syncthreads();
  float sum = 0.f;
#pragma unroll
  for (int i = 0; i < 8; ++i) { vals[i] = __expf(vals[i] - m); sum += vals[i]; }
  red[tid] = sum; __syncthreads();
  for (int off = 128; off > 0; off >>= 1) {
    if (tid < off) red[tid] += red[tid + off];
    __syncthreads();
  }
  const float inv = 1.0f / red[0];
#pragma unroll
  for (int i = 0; i < 8; ++i) {
    float pv = vals[i] * inv;
    s[tid + i * 256] = pv;          // fp32 attention output
    p[tid + i * 256] = (__bf16)pv;  // bf16 copy for P@v WMMA
  }
}

// ---------------- launch ----------------
extern "C" void kernel_launch(void* const* d_in, const int* in_sizes, int n_in,
                              void* d_out, int out_size, void* d_ws, size_t ws_size,
                              hipStream_t stream) {
  const float* q    = (const float*)d_in[0];
  const float* k    = (const float*)d_in[1];
  const float* v    = (const float*)d_in[2];
  const int*   mask = (const int*)d_in[3];
  const float* w    = (const float*)d_in[4];

  float* out  = (float*)d_out;                                  // B*TQ*D
  float* attn = out + (long)ATT_B * ATT_TQ * ATT_D;             // B*TQ*TK

  const size_t nQ = (size_t)ATT_B * ATT_TQ * ATT_D;   // 8,388,608

  __bf16* qb  = (__bf16*)d_ws;       // bf16 q              (B,TQ,D)
  __bf16* kb  = qb  + nQ;            // bf16 k              (B,TK,D)
  __bf16* vtb = kb  + nQ;            // bf16 v^T            (B,D,TK)
  __bf16* wtb = vtb + nQ;            // bf16 W^T            (D,D)
  __bf16* qwb = wtb + (size_t)ATT_D * ATT_D;  // bf16 qW    (B,TQ,D)
  __bf16* pb  = qwb + nQ;            // bf16 attention      (B,TQ,TK)

  // 1) precision / layout conversions
  cvt_f32_to_bf16<<<4096, 256, 0, stream>>>(q, qb, (long)nQ);
  cvt_f32_to_bf16<<<4096, 256, 0, stream>>>(k, kb, (long)nQ);
  transpose_f32_to_bf16<<<dim3((ATT_TK * ATT_D) / 256, 1, ATT_B), 256, 0, stream>>>(
      v, vtb, ATT_TK, ATT_D);
  transpose_f32_to_bf16<<<dim3((ATT_D * ATT_D) / 256, 1, 1), 256, 0, stream>>>(
      w, wtb, ATT_D, ATT_D);

  // 2) qw = q @ W   (fold batch into M: 16384 x 512 x 512), bf16 result only
  wmma_gemm_bf16<false, false, true>
      <<<dim3(ATT_D / 128, (ATT_B * ATT_TQ) / 128, 1), 256, 0, stream>>>(
      qb, wtb, nullptr, qwb, nullptr,
      ATT_D, ATT_D, ATT_D, ATT_D, 0, 0, 0, 0);

  // 3) scores = qw @ k^T, fused mask -> -inf, fp32 into attention output
  wmma_gemm_bf16<true, true, false>
      <<<dim3(ATT_TK / 128, ATT_TQ / 128, ATT_B), 256, 0, stream>>>(
      qwb, kb, attn, nullptr, mask,
      ATT_D, ATT_D, ATT_D, ATT_TK,
      (long long)ATT_TQ * ATT_D, (long long)ATT_TK * ATT_D,
      (long long)ATT_TQ * ATT_TK, (long long)ATT_TQ * ATT_TK);

  // 4) softmax rows: fp32 in-place (attention output) + bf16 copy for P@v
  softmax_rows<<<ATT_B * ATT_TQ, 256, 0, stream>>>(attn, pb);

  // 5) out = P @ v  (A = P bf16, BT = v^T bf16)
  wmma_gemm_bf16<false, true, false>
      <<<dim3(ATT_D / 128, ATT_TQ / 128, ATT_B), 256, 0, stream>>>(
      pb, vtb, out, nullptr, nullptr,
      ATT_TK, ATT_TK, ATT_TK, ATT_D,
      (long long)ATT_TQ * ATT_TK, (long long)ATT_D * ATT_TK,
      (long long)ATT_TQ * ATT_D, 0);
}
